// Attention_23656679867685
// MI455X (gfx1250) — matrix operand order repaired and checked by reading, not compiled, not run on hardware
//
#include <hip/hip_runtime.h>
#include <hip/hip_bf16.h>

typedef _Float16 half_t;
typedef __attribute__((ext_vector_type(16))) _Float16 v16h;
typedef __attribute__((ext_vector_type(8)))  _Float16 v8h;
typedef __attribute__((ext_vector_type(8)))  float    v8f;

#define BB   2
#define NN   2048
#define DD   1024
#define HH   16
#define HDIM 64
#define MTOK (BB*NN)      /* 4096 tokens */
#define SM_SCALE 0.125f   /* 1/sqrt(64) */
#define RMS_EPS 1e-6f
#define MINIT (-3.0e38f)

// ---------------------------------------------------------------------------
// WMMA helpers: exact CDNA5 16-bit fragment layouts (ISA 7.12.2), wave32.
// ---------------------------------------------------------------------------
__device__ __forceinline__ v8f wmma16(v16h a, v16h b, v8f c) {
  // v_wmma_f32_16x16x32_f16, fp32 accumulate
  return __builtin_amdgcn_wmma_f32_16x16x32_f16(false, a, false, b,
                                                (short)0, c, false, false);
}

// A fragment: 16x32 f16 tile from row-major [M x lda] at (row0,k0)=base.
// lanes 0-15: row=lane, K in {0..7, 16..23}; lanes 16-31: row=lane-16,
// K in {8..15, 24..31}. Two contiguous 16B loads per lane.
__device__ __forceinline__ v16h load_a_frag(const half_t* base, int lda, int lane) {
  const int g   = (lane >> 4) & 1;
  const int row = lane & 15;
  const half_t* p = base + (size_t)row * lda + g * 8;
  v8h lo8 = *(const v8h*)(p);       // K = 8g .. 8g+7
  v8h hi8 = *(const v8h*)(p + 16);  // K = 8g+16 .. 8g+23
  v16h a;
#pragma unroll
  for (int i = 0; i < 8; ++i) { a[i] = lo8[i]; a[8 + i] = hi8[i]; }
  return a;
}

// B fragment: 32x16 f16 (KxN) where B[k][n] = Bt[n][k], Bt row-major [N x ldb]
// at (n0,k0)=bt. lanes 0-15: n=lane, K in {0..7, 16..23}; lanes 16-31: n=lane-16, K in {8..15, 24..31}.
__device__ __forceinline__ v16h load_b_frag(const half_t* bt, int ldb, int lane) {
  const int g = (lane >> 4) & 1;
  const int n = lane & 15;
  const half_t* p = bt + (size_t)n * ldb + g * 8;
  v8h lo8 = *(const v8h*)(p);
  v8h hi8 = *(const v8h*)(p + 16);
  v16h b;
#pragma unroll
  for (int i = 0; i < 8; ++i) { b[i] = lo8[i]; b[8 + i] = hi8[i]; }
  return b;
}

__device__ __forceinline__ float redmax16(float v) {
#pragma unroll
  for (int m = 1; m < 16; m <<= 1) v = fmaxf(v, __shfl_xor(v, m, 32));
  return v;
}
__device__ __forceinline__ float redsum16(float v) {
#pragma unroll
  for (int m = 1; m < 16; m <<= 1) v += __shfl_xor(v, m, 32);
  return v;
}

// ---------------------------------------------------------------------------
// GEMM: C[M,N] = A[M,K](f16) * Bt[N,K]^T(f16) + bias[N], f32 out.
// 256 threads = 8 waves; block tile 128x128; wave tile 32x64; K-step 32.
// Per K-step per wave: 2 A-frags + 4 B-frags (12 x b128) feed 8 WMMAs.
// ---------------------------------------------------------------------------
__global__ __launch_bounds__(256) void gemm_f16wmma(
    const half_t* __restrict__ A, const half_t* __restrict__ Bt,
    const float* __restrict__ bias, float* __restrict__ C,
    int M, int N, int K) {
  const int lane = threadIdx.x & 31;
  const int wave = threadIdx.x >> 5;
  const int m0 = blockIdx.x * 128 + (wave >> 1) * 32;
  const int n0 = blockIdx.y * 128 + (wave & 1) * 64;

  v8f acc[2][4] = {};
  for (int k0 = 0; k0 < K; k0 += 32) {
    v16h a0 = load_a_frag(A + (size_t)(m0 +  0) * K + k0, K, lane);
    v16h a1 = load_a_frag(A + (size_t)(m0 + 16) * K + k0, K, lane);
    v16h b0 = load_b_frag(Bt + (size_t)(n0 +  0) * K + k0, K, lane);
    v16h b1 = load_b_frag(Bt + (size_t)(n0 + 16) * K + k0, K, lane);
    v16h b2 = load_b_frag(Bt + (size_t)(n0 + 32) * K + k0, K, lane);
    v16h b3 = load_b_frag(Bt + (size_t)(n0 + 48) * K + k0, K, lane);
    acc[0][0] = wmma16(a0, b0, acc[0][0]);
    acc[0][1] = wmma16(a0, b1, acc[0][1]);
    acc[0][2] = wmma16(a0, b2, acc[0][2]);
    acc[0][3] = wmma16(a0, b3, acc[0][3]);
    acc[1][0] = wmma16(a1, b0, acc[1][0]);
    acc[1][1] = wmma16(a1, b1, acc[1][1]);
    acc[1][2] = wmma16(a1, b2, acc[1][2]);
    acc[1][3] = wmma16(a1, b3, acc[1][3]);
  }
  // C/D layout: VGPR r -> row r (lanes 0-15) / row 8+r (lanes 16-31), col=lane&15
  const int col = lane & 15;
#pragma unroll
  for (int i = 0; i < 2; ++i) {
    const int rbase = m0 + 16 * i + ((lane < 16) ? 0 : 8);
#pragma unroll
    for (int j = 0; j < 4; ++j) {
      const int c = n0 + 16 * j + col;
      const float bvv = bias[c];
#pragma unroll
      for (int r = 0; r < 8; ++r)
        C[(size_t)(rbase + r) * N + c] = acc[i][j][r] + bvv;
    }
  }
}

// ---------------------------------------------------------------------------
// Elementwise conversion kernels
// ---------------------------------------------------------------------------
__global__ void cvt_f32_to_f16(const float* __restrict__ in,
                               half_t* __restrict__ out, int n) {
  int i = blockIdx.x * blockDim.x + threadIdx.x;
  if (i < n) out[i] = (half_t)in[i];
}

// Wt[n][k] = (f16) W[k][n]  (so GEMM B-fragments are contiguous per lane)
__global__ void transpose_w_f16(const float* __restrict__ W,
                                half_t* __restrict__ Wt) {
  int i = blockIdx.x * blockDim.x + threadIdx.x;  // i = n*DD + k
  int k = i & (DD - 1);
  int n = i >> 10;
  Wt[i] = (half_t)W[(size_t)k * DD + n];
}

// ---------------------------------------------------------------------------
// RMSNorm(head_dim) + RoPE, f32 in [B,N,D] -> f16 out [B,H,N,HD]
// one wave32 per (b,n,h) row; each lane owns an interleaved (even,odd) pair
// ---------------------------------------------------------------------------
__global__ __launch_bounds__(256) void qk_norm_rope(
    const float* __restrict__ X, const float* __restrict__ w,
    const float* __restrict__ freqs, half_t* __restrict__ out) {
  const int lane = threadIdx.x & 31;
  const int rowid = blockIdx.x * 8 + (threadIdx.x >> 5);  // b*NN*HH + n*HH + h
  const int h = rowid % HH;
  const int t = rowid / HH;
  const int n = t % NN;
  const int b = t / NN;

  const float* src = X + ((size_t)(b * NN + n)) * DD + h * HDIM;
  const float x0 = src[2 * lane];
  const float x1 = src[2 * lane + 1];
  float ss = x0 * x0 + x1 * x1;
#pragma unroll
  for (int m = 1; m < 32; m <<= 1) ss += __shfl_xor(ss, m, 32);
  const float inv = rsqrtf(ss * (1.0f / HDIM) + RMS_EPS);

  const float t0 = x0 * inv * w[2 * lane];
  const float t1 = x1 * inv * w[2 * lane + 1];
  const float f0 = freqs[(size_t)n * HDIM + 2 * lane];
  const float f1 = freqs[(size_t)n * HDIM + 2 * lane + 1];
  // out[2i] = t0*cos(f0) - t1*sin(f0);  out[2i+1] = t1*cos(f1) + t0*sin(f1)
  const float o0 = t0 * __cosf(f0) - t1 * __sinf(f0);
  const float o1 = t1 * __cosf(f1) + t0 * __sinf(f1);

  half_t* dst = out + (((size_t)(b * HH + h)) * NN + n) * HDIM + 2 * lane;
  dst[0] = (half_t)o0;
  dst[1] = (half_t)o1;
}

// V: f32 [B,N,D] -> f16 [B,H,N,HD]
__global__ void v_relayout(const float* __restrict__ V32,
                           half_t* __restrict__ Vh) {
  int i = blockIdx.x * blockDim.x + threadIdx.x;  // ((b*HH+h)*NN+n)*HDIM+e
  int e = i & (HDIM - 1);
  int r = i >> 6;
  int n = r % NN;
  int r2 = r / NN;
  int h = r2 % HH;
  int b = r2 / HH;
  Vh[i] = (half_t)V32[((size_t)(b * NN + n)) * DD + h * HDIM + e];
}

// ---------------------------------------------------------------------------
// Flash attention: grid (N/64, B*H), 128 threads = 4 waves; each wave owns
// 16 queries, loops over keys in chunks of 32. Scores & PV on WMMA f16,
// online softmax in fp32. Output written token-major [4096, 1024] f16.
// ---------------------------------------------------------------------------
__global__ __launch_bounds__(128) void flash_attn(
    const half_t* __restrict__ Qh, const half_t* __restrict__ Kh,
    const half_t* __restrict__ Vh, const unsigned char* __restrict__ mask,
    half_t* __restrict__ Oh) {
  const int lane = threadIdx.x & 31;
  const int wave = threadIdx.x >> 5;
  const int bh = blockIdx.y;
  const int b = bh / HH;
  const int h = bh % HH;
  const int q0 = blockIdx.x * 64 + wave * 16;

  const half_t* Qb = Qh + (size_t)bh * NN * HDIM;
  const half_t* Kb = Kh + (size_t)bh * NN * HDIM;
  const half_t* Vb = Vh + (size_t)bh * NN * HDIM;
  const unsigned char* mb = mask + (size_t)b * NN;

  __shared__ __align__(16) half_t Vt[HDIM][32];   // transposed V chunk [hd][key]
  __shared__ __align__(16) half_t Ps[4][16][32];  // per-wave probs [q][key]

  // Q fragments (K-dim = head channels, two 32-chunks)
  const v16h qa0 = load_a_frag(Qb + (size_t)q0 * HDIM + 0, HDIM, lane);
  const v16h qa1 = load_a_frag(Qb + (size_t)q0 * HDIM + 32, HDIM, lane);

  v8f o0 = {}, o1 = {}, o2 = {}, o3 = {};
  float mrow[8], lrow[8];
#pragma unroll
  for (int r = 0; r < 8; ++r) { mrow[r] = MINIT; lrow[r] = 0.0f; }

  for (int kk = 0; kk < NN; kk += 32) {
    __syncthreads();
    {  // cooperative transpose of V chunk [32 keys x 64 hd] -> Vt[hd][key]
      const int key = threadIdx.x >> 2;
      const int part = threadIdx.x & 3;
      const half_t* src = Vb + (size_t)(kk + key) * HDIM + part * 16;
      v8h u0 = *(const v8h*)src;
      v8h u1 = *(const v8h*)(src + 8);
#pragma unroll
      for (int i = 0; i < 8; ++i) {
        Vt[part * 16 + i][key] = u0[i];
        Vt[part * 16 + 8 + i][key] = u1[i];
      }
    }
    __syncthreads();

    // S = Q (16x64) * K^T (64x32) : two 16x16 score tiles, each 2 WMMAs
    v8f s0 = {}, s1 = {};
    s0 = wmma16(qa0, load_b_frag(Kb + (size_t)(kk + 0) * HDIM + 0, HDIM, lane), s0);
    s0 = wmma16(qa1, load_b_frag(Kb + (size_t)(kk + 0) * HDIM + 32, HDIM, lane), s0);
    s1 = wmma16(qa0, load_b_frag(Kb + (size_t)(kk + 16) * HDIM + 0, HDIM, lane), s1);
    s1 = wmma16(qa1, load_b_frag(Kb + (size_t)(kk + 16) * HDIM + 32, HDIM, lane), s1);

    const bool ok0 = mb[kk + (lane & 15)] != 0;
    const bool ok1 = mb[kk + 16 + (lane & 15)] != 0;
    const int prow_base = (lane < 16) ? 0 : 8;
    const int pcol = lane & 15;

#pragma unroll
    for (int r = 0; r < 8; ++r) {
      const float a0 = ok0 ? s0[r] * SM_SCALE : -1e30f;
      const float a1 = ok1 ? s1[r] * SM_SCALE : -1e30f;
      const float cmax = redmax16(fmaxf(a0, a1));
      const float mnew = fmaxf(mrow[r], cmax);
      const float corr = __expf(mrow[r] - mnew);
      const float p0 = __expf(a0 - mnew);
      const float p1 = __expf(a1 - mnew);
      lrow[r] = lrow[r] * corr + redsum16(p0 + p1);
      mrow[r] = mnew;
      o0[r] *= corr; o1[r] *= corr; o2[r] *= corr; o3[r] *= corr;
      Ps[wave][prow_base + r][pcol] = (half_t)p0;
      Ps[wave][prow_base + r][pcol + 16] = (half_t)p1;
    }
    // wave-private LDS RAW: drain DS counter (CDNA5 split counters)
    asm volatile("s_wait_dscnt 0" ::: "memory");
    __builtin_amdgcn_wave_barrier();

    // O += P (16x32) * V (32x64) : 4 WMMAs, B-frags from transposed LDS
    const v16h pa = load_a_frag(&Ps[wave][0][0], 32, lane);
    o0 = wmma16(pa, load_b_frag(&Vt[0][0], 32, lane), o0);
    o1 = wmma16(pa, load_b_frag(&Vt[16][0], 32, lane), o1);
    o2 = wmma16(pa, load_b_frag(&Vt[32][0], 32, lane), o2);
    o3 = wmma16(pa, load_b_frag(&Vt[48][0], 32, lane), o3);
  }

  // finalize: divide by row sums, write token-major f16 [4096, 1024]
  const int col = lane & 15;
  v8f oo[4] = {o0, o1, o2, o3};
#pragma unroll
  for (int r = 0; r < 8; ++r) {
    const int token = q0 + ((lane < 16) ? r : 8 + r);
    const float linv = 1.0f / lrow[r];
    half_t* dst = Oh + (size_t)(b * NN + token) * DD + h * HDIM;
#pragma unroll
    for (int j = 0; j < 4; ++j)
      dst[16 * j + col] = (half_t)(oo[j][r] * linv);
  }
}

// zero rows where mask is false
__global__ void mask_zero(float* __restrict__ out,
                          const unsigned char* __restrict__ mask) {
  int i = blockIdx.x * blockDim.x + threadIdx.x;  // token*DD + c
  int token = i >> 10;
  int b = token / NN;
  int n = token % NN;
  if (!mask[(size_t)b * NN + n]) out[i] = 0.0f;
}

// ---------------------------------------------------------------------------
extern "C" void kernel_launch(void* const* d_in, const int* in_sizes, int n_in,
                              void* d_out, int out_size, void* d_ws, size_t ws_size,
                              hipStream_t stream) {
  const float* x     = (const float*)d_in[0];
  const unsigned char* mask = (const unsigned char*)d_in[1];  // jax bool = i8
  const float* freqs = (const float*)d_in[2];
  const float* Wq = (const float*)d_in[3];
  const float* bq = (const float*)d_in[4];
  const float* Wk = (const float*)d_in[5];
  const float* bk = (const float*)d_in[6];
  const float* Wv = (const float*)d_in[7];
  const float* bv = (const float*)d_in[8];
  const float* Wo = (const float*)d_in[9];
  const float* bo = (const float*)d_in[10];
  const float* qw = (const float*)d_in[11];
  const float* kw = (const float*)d_in[12];
  float* out = (float*)d_out;

  char* ws = (char*)d_ws;
  const size_t SZ_XH = (size_t)MTOK * DD * sizeof(half_t);   // 8 MiB
  const size_t SZ_WT = (size_t)DD * DD * sizeof(half_t);     // 2 MiB
  const size_t SZ_F32 = (size_t)MTOK * DD * sizeof(float);   // 16 MiB
  const size_t SZ_H16 = (size_t)MTOK * DD * sizeof(half_t);  // 8 MiB

  half_t* Xh  = (half_t*)ws;              ws += SZ_XH;
  half_t* WqT = (half_t*)ws;              ws += SZ_WT;
  half_t* WkT = (half_t*)ws;              ws += SZ_WT;
  half_t* WvT = (half_t*)ws;              ws += SZ_WT;
  half_t* WoT = (half_t*)ws;              ws += SZ_WT;
  float*  Q32 = (float*)ws;               ws += SZ_F32;
  float*  K32 = (float*)ws;               ws += SZ_F32;
  float*  V32 = (float*)ws;               ws += SZ_F32;
  half_t* Qh  = (half_t*)ws;              ws += SZ_H16;
  half_t* Kh  = (half_t*)ws;              ws += SZ_H16;
  half_t* Vh  = (half_t*)ws;              ws += SZ_H16;
  half_t* Oh  = (half_t*)ws;              ws += SZ_H16;
  (void)ws_size; (void)n_in; (void)in_sizes; (void)out_size;

  // 1) f16 staging
  cvt_f32_to_f16<<<(MTOK * DD) / 256, 256, 0, stream>>>(x, Xh, MTOK * DD);
  transpose_w_f16<<<(DD * DD) / 256, 256, 0, stream>>>(Wq, WqT);
  transpose_w_f16<<<(DD * DD) / 256, 256, 0, stream>>>(Wk, WkT);
  transpose_w_f16<<<(DD * DD) / 256, 256, 0, stream>>>(Wv, WvT);
  transpose_w_f16<<<(DD * DD) / 256, 256, 0, stream>>>(Wo, WoT);

  // 2) QKV projections (WMMA): block tile 128x128
  dim3 ggrid(MTOK / 128, DD / 128);
  gemm_f16wmma<<<ggrid, 256, 0, stream>>>(Xh, WqT, bq, Q32, MTOK, DD, DD);
  gemm_f16wmma<<<ggrid, 256, 0, stream>>>(Xh, WkT, bk, K32, MTOK, DD, DD);
  gemm_f16wmma<<<ggrid, 256, 0, stream>>>(Xh, WvT, bv, V32, MTOK, DD, DD);

  // 3) head RMSNorm + RoPE -> f16 [B,H,N,HD]; V relayout
  const int nrows = BB * NN * HH;  // 65536
  qk_norm_rope<<<nrows / 8, 256, 0, stream>>>(Q32, qw, freqs, Qh);
  qk_norm_rope<<<nrows / 8, 256, 0, stream>>>(K32, kw, freqs, Kh);
  v_relayout<<<(MTOK * DD) / 256, 256, 0, stream>>>(V32, Vh);

  // 4) flash attention (WMMA)
  dim3 agrid(NN / 64, BB * HH);
  flash_attn<<<agrid, 128, 0, stream>>>(Qh, Kh, Vh, mask, Oh);

  // 5) output projection (WMMA) + masking
  gemm_f16wmma<<<ggrid, 256, 0, stream>>>(Oh, WoT, bo, out, MTOK, DD, DD);
  mask_zero<<<(MTOK * DD) / 256, 256, 0, stream>>>(out, mask);
}
